// KNRM_89902255440779
// MI455X (gfx1250) — matrix operand order repaired and checked by reading, not compiled, not run on hardware
//
#include <hip/hip_runtime.h>
#include <math.h>

#define K_KERNELS 11
#define BQ 32
#define BD 512
#define EDIM 300
#define EP 320            // padded E, multiple of 32 (WMMA K-step)
#define DC 64             // docs per chunk
#define NCHUNK (BD / DC)  // 8
#define NTHREADS 256
#define NWAVES 8

typedef __attribute__((ext_vector_type(16))) _Float16 v16h;
typedef __attribute__((ext_vector_type(8)))  _Float16 v8h;
typedef __attribute__((ext_vector_type(8)))  float    v8f;

struct KParams {
    float mu[K_KERNELS];
    float inv2s2[K_KERNELS];
};

// Gather embedding rows by token id, L2-normalize, store as f16 into LDS
// (padded to EP with zeros). One wave per row, lanes stride the E dimension.
__device__ __forceinline__ void gather_normalize_rows(
    const int* __restrict__ tok, int nrows,
    const float* __restrict__ emb,
    _Float16* __restrict__ dst, float* __restrict__ mask,
    int wave, int lane)
{
    for (int r = wave; r < nrows; r += NWAVES) {
        int token = tok[r];
        float m = (token > 0) ? 1.0f : 0.0f;
        const float* row = emb + (long long)token * EDIM;
        float v[EP / 32];
        float ss = 0.0f;
        #pragma unroll
        for (int i = 0; i < EP / 32; ++i) {
            int e = lane + i * 32;
            float x = (e < EDIM) ? row[e] : 0.0f;
            v[i] = x;
            ss += x * x;
        }
        #pragma unroll
        for (int off = 16; off > 0; off >>= 1)
            ss += __shfl_xor(ss, off, 32);
        float rn = 1.0f / (sqrtf(ss) + 1e-13f);
        #pragma unroll
        for (int i = 0; i < EP / 32; ++i)
            dst[r * EP + lane + i * 32] = (_Float16)(v[i] * rn);
        if (lane == 0) mask[r] = m;
    }
}

__global__ __launch_bounds__(NTHREADS)
void knrm_fused_kernel(const int* __restrict__ qtok,
                       const int* __restrict__ dtok,
                       const float* __restrict__ emb,
                       const float* __restrict__ w,
                       const float* __restrict__ bias,
                       float* __restrict__ out,
                       KParams P)
{
    __shared__ __align__(32) _Float16 s_q[BQ * EP];   // 20 KB
    __shared__ __align__(32) _Float16 s_d[DC * EP];   // 40 KB
    __shared__ float s_qm[BQ];
    __shared__ float s_dm[DC];
    __shared__ float s_pkq[BQ][K_KERNELS + 1];        // padded stride

    const int b    = blockIdx.x;
    const int tid  = threadIdx.x;
    const int lane = tid & 31;
    const int wave = tid >> 5;

    for (int i = tid; i < BQ * (K_KERNELS + 1); i += NTHREADS)
        ((float*)s_pkq)[i] = 0.0f;

    // Phase 1: query embeddings -> LDS (f16, normalized)
    gather_normalize_rows(qtok + b * BQ, BQ, emb, s_q, s_qm, wave, lane);
    __syncthreads();

    // Tile ownership: wave -> (query-tile, doc-tile) of a 32x64 cosine block
    const int qt = wave & 1;   // 2 query tiles of 16
    const int dt = wave >> 1;  // 4 doc tiles of 16
    const int g  = lane >> 4;  // half-wave group (WMMA lane striping)
    const int ln = lane & 15;

    // Query masks for this lane's 8 C-matrix rows are chunk-invariant: hoist.
    float qmr[8];
    #pragma unroll
    for (int r = 0; r < 8; ++r)
        qmr[r] = s_qm[qt * 16 + g * 8 + r];

    // Per-lane kernel-pooling accumulators, summed over ALL doc chunks; the
    // cross-lane (16 doc columns) reduction happens once at the end.
    float accum[8][K_KERNELS];
    #pragma unroll
    for (int r = 0; r < 8; ++r)
        #pragma unroll
        for (int k = 0; k < K_KERNELS; ++k)
            accum[r][k] = 0.0f;

    for (int c = 0; c < NCHUNK; ++c) {
        __syncthreads();  // previous chunk's pooling done before s_d overwrite
        gather_normalize_rows(dtok + b * BD + c * DC, DC, emb, s_d, s_dm,
                              wave, lane);
        __syncthreads();

        // Prefetch next chunk's doc embedding rows into cache while this
        // chunk computes (global_prefetch_b8; rows are L2-resident).
        if (c + 1 < NCHUNK) {
            const int* ntok = dtok + b * BD + (c + 1) * DC;
            for (int r = wave; r < DC; r += NWAVES) {
                int token = ntok[r];
                const float* row = emb + (long long)token * EDIM;
                __builtin_prefetch(row + lane * 10, 0, 3);  // 40B stripes
            }
        }

        // 16x16 cosine tile: D = A(16xK) * B(Kx16), K stepped by 32, f32 acc.
        v8f acc = {};
        const _Float16* qbase = &s_q[(qt * 16 + ln) * EP];
        const _Float16* dbase = &s_d[(dt * 16 + ln) * EP + g * 16];
        #pragma unroll
        for (int ek = 0; ek < EP; ek += 32) {
            // A fragment (16x32 f16): lane holds row M=ln; half h in [0,16)
            // carries K = g*16 + h.
            v8h lo = *(const v8h*)(qbase + ek + g * 16);
            v8h hi = *(const v8h*)(qbase + ek + g * 16 + 8);
            v16h a = __builtin_shufflevector(
                lo, hi, 0, 1, 2, 3, 4, 5, 6, 7, 8, 9, 10, 11, 12, 13, 14, 15);
            // B fragment (32x16 f16): lane holds column N=ln, halfs carry
            // K = g*16 + 0..15 -> one contiguous 32B LDS read per lane.
            v16h bm = *(const v16h*)(dbase + ek);
            acc = __builtin_amdgcn_wmma_f32_16x16x32_f16(
                false, a, false, bm, (short)0, acc, false, false);
        }

        // Gaussian kernel pooling over this tile, accumulated in registers.
        // C layout: VGPR r holds M = r + 8*g, N = ln.
        float dm = s_dm[dt * 16 + ln];
        #pragma unroll
        for (int r = 0; r < 8; ++r) {
            float m  = qmr[r] * dm;
            float cv = acc[r] * m;
            #pragma unroll
            for (int k = 0; k < K_KERNELS; ++k) {
                float t = cv - P.mu[k];
                accum[r][k] += m * __expf(-t * t * P.inv2s2[k]);
            }
        }
    }

    // One cross-lane reduction over the 16 doc columns, then LDS atomics to
    // combine the 4 doc tiles (waves) into per-(q,k) sums.
    #pragma unroll
    for (int r = 0; r < 8; ++r) {
        int q = qt * 16 + g * 8 + r;
        #pragma unroll
        for (int k = 0; k < K_KERNELS; ++k) {
            float v = accum[r][k];
            v += __shfl_xor(v, 1, 32);
            v += __shfl_xor(v, 2, 32);
            v += __shfl_xor(v, 4, 32);
            v += __shfl_xor(v, 8, 32);
            if (ln == 0) atomicAdd(&s_pkq[q][k], v);  // ds_add_f32
        }
    }

    __syncthreads();

    // Epilogue: log/clip/mask, fold in (per_kernel @ w + b), reduce over Q.
    if (wave == 0) {
        int q = lane;  // BQ == 32 == wave width
        float qm = s_qm[q];
        float accq = 0.0f;
        #pragma unroll
        for (int k = 0; k < K_KERNELS; ++k) {
            float v  = s_pkq[q][k];
            float lp = __logf(fmaxf(v, 1e-10f)) * qm;
            accq += lp * w[k];
        }
        #pragma unroll
        for (int off = 16; off > 0; off >>= 1)
            accq += __shfl_xor(accq, off, 32);
        if (lane == 0) out[b] = accq + bias[0];
    }
}

extern "C" void kernel_launch(void* const* d_in, const int* in_sizes, int n_in,
                              void* d_out, int out_size, void* d_ws, size_t ws_size,
                              hipStream_t stream) {
    (void)in_sizes; (void)n_in; (void)d_ws; (void)ws_size; (void)out_size;
    const int*   qtok = (const int*)d_in[0];    // [256,32]
    const int*   dtok = (const int*)d_in[1];    // [256,512]
    const float* emb  = (const float*)d_in[2];  // [100000,300]
    const float* w    = (const float*)d_in[3];  // [11]
    const float* bias = (const float*)d_in[4];  // [1]
    float*       out  = (float*)d_out;          // [256]

    // Kernel params exactly per the reference's _kernel_params(11)
    KParams P;
    const float bin = 2.0f / (K_KERNELS - 1);           // 0.2
    P.mu[0] = 1.0f;
    P.mu[1] = 1.0f - bin * 0.5f;                        // 0.9
    for (int i = 1; i < K_KERNELS - 1; ++i)
        P.mu[i + 1] = P.mu[i] - bin;                    // 0.7 ... -0.9
    float sig[K_KERNELS];
    sig[0] = 1e-4f;
    for (int k = 1; k < K_KERNELS; ++k) sig[k] = 0.5f * bin;  // 0.1
    for (int k = 0; k < K_KERNELS; ++k)
        P.inv2s2[k] = 1.0f / (2.0f * sig[k] * sig[k]);

    knrm_fused_kernel<<<256, NTHREADS, 0, stream>>>(qtok, dtok, emb, w, bias,
                                                    out, P);
}
